// Transition_85212151153187
// MI455X (gfx1250) — matrix operand order repaired and checked by reading, not compiled, not run on hardware
//
#include <hip/hip_runtime.h>
#include <hip/hip_bf16.h>

// ---------------------------------------------------------------------------
// Persistent-WMMA LSTM scan for MI455X (gfx1250, wave32).
//   - f16 weights/activations, f32 accumulation via v_wmma_f32_16x16x32_f16
//   - weights pre-swizzled into per-lane A tiles (contiguous k = 16*half + j)
//   - h broadcast into all 16 B columns -> D columns identical (free extract)
//   - 64 persistent WGs, one device barrier per timestep, double-buffered h
// ---------------------------------------------------------------------------

typedef __attribute__((ext_vector_type(16))) _Float16 v16h;
typedef __attribute__((ext_vector_type(8)))  float    v8f;

#define HIDN     512
#define TSTEPS   32768
#define NWG      64
#define TPB      128

// workspace layout (bytes)
#define WS_CNT    0                       // 1 int barrier counter
#define WS_H16    64                      // 2 slots * 2 lstm * 512 f16 = 4096 B
#define WS_X16    4224                    // 32768 * 32 f16 = 2097152 B
#define WS_BIAS   (4224 + 2097152)        // 2*2048 f32 (b_ih+b_hh) = 16384 B
#define WS_ATILE  (WS_BIAS + 16384)       // 64*4*17*32*16 f16 = 4456448 B

__device__ __forceinline__ float sigmoidf_(float x) {
    return 1.0f / (1.0f + __expf(-x));
}

// ---------------- setup kernels (all re-run each launch; graph-safe) -------

// d_out <- head biases; zero h double-buffer; zero barrier counter
__global__ void k_init(float* __restrict__ out,
                       const float* __restrict__ bxyz, const float* __restrict__ bzeta,
                       const float* __restrict__ buvw, const float* __restrict__ bpqr,
                       char* __restrict__ ws) {
    int idx = blockIdx.x * blockDim.x + threadIdx.x;
    if (idx < TSTEPS * 12) {
        int j = idx % 12;
        float v = (j < 3) ? bxyz[j] : (j < 6) ? bzeta[j - 3]
                : (j < 9) ? buvw[j - 6] : bpqr[j - 9];
        out[idx] = v;
    }
    if (idx < 2048) ((_Float16*)(ws + WS_H16))[idx] = (_Float16)0.0f;
    if (idx == 0)   *((int*)(ws + WS_CNT)) = 0;
}

// x_f16[t][0..31]: 19 inputs, zero-padded to a 32-wide K chunk
__global__ void k_x16(const float* __restrict__ sa, char* __restrict__ ws) {
    int idx = blockIdx.x * blockDim.x + threadIdx.x;
    if (idx >= TSTEPS * 32) return;
    int col = idx & 31, t = idx >> 5;
    float v = (col < 19) ? sa[t * 19 + col] : 0.0f;
    ((_Float16*)(ws + WS_X16))[idx] = (_Float16)v;
}

// fused gate bias (kept f32 for accuracy)
__global__ void k_bias(const float* __restrict__ bih1, const float* __restrict__ bhh1,
                       const float* __restrict__ bih2, const float* __restrict__ bhh2,
                       char* __restrict__ ws) {
    int idx = blockIdx.x * blockDim.x + threadIdx.x;
    if (idx >= 2 * 2048) return;
    int l = idx >> 11, r = idx & 2047;
    ((float*)(ws + WS_BIAS))[idx] = l ? (bih2[r] + bhh2[r]) : (bih1[r] + bhh1[r]);
}

// Weight tiles swizzled into per-lane 16x32 A tiles (contiguous k order, same as B):
// flat idx = (((wg*4 + gate)*17 + kchunk)*32 + lane)*16 + j
// lane = M row (mod 16); lane>=16 selects the K=16..31 half.
__global__ void k_atile(const float* __restrict__ Wih1, const float* __restrict__ Whh1,
                        const float* __restrict__ Wih2, const float* __restrict__ Whh2,
                        char* __restrict__ ws) {
    int idx = blockIdx.x * blockDim.x + threadIdx.x;
    const int NTOT = NWG * 4 * 17 * 32 * 16;           // 2,228,224
    if (idx >= NTOT) return;
    int j    = idx & 15;
    int lane = (idx >> 4) & 31;
    int rem  = idx >> 9;                               // (wg*4+q)*17 + c
    int c    = rem % 17;
    int tf   = rem / 17;
    int q    = tf & 3;
    int g    = tf >> 2;
    int lstm = g >> 5;
    int ub   = (g & 31) * 16;
    int m    = lane & 15, sel = lane >> 4;
    int row  = q * 512 + ub + m;                       // gate-major row in W
    int kin  = sel * 16 + j;
    int k    = c * 32 + kin;                           // 0..543
    const float* Whh = lstm ? Whh2 : Whh1;
    const float* Wih = lstm ? Wih2 : Wih1;
    float v;
    if (k < 512) v = Whh[row * 512 + k];
    else { int kp = k - 512; v = (kp < 19) ? Wih[row * 19 + kp] : 0.0f; }
    ((_Float16*)(ws + WS_ATILE))[idx] = (_Float16)v;
}

// ---------------- persistent recurrent kernel ------------------------------
// WG g: lstm = g>>5, 16 hidden units [ub, ub+16). Wave w computes gate w for
// those 16 units: 17 chained WMMAs over K=544 (512 h + 32 x).
__global__ void __launch_bounds__(TPB) k_lstm(const float* __restrict__ Wxyz,
                                              const float* __restrict__ Wzeta,
                                              const float* __restrict__ Wuvw,
                                              const float* __restrict__ Wpqr,
                                              float* __restrict__ out,
                                              char* __restrict__ ws) {
    extern __shared__ __align__(32) char smem[];
    v16h*  Alds  = (v16h*)smem;                        // 4*17*32 tiles = 69632 B
    float* biasL = (float*)(smem + 69632);             // 64 gate-row biases
    float* gbuf  = (float*)(smem + 69888);             // [4 gates][16 units]
    float* hsf   = (float*)(smem + 70144);             // h (f32) for head dots
    float* cst   = (float*)(smem + 70208);             // c state
    float* hw    = (float*)(smem + 70272);             // [6][16] head weights

    const int tid  = threadIdx.x;
    const int g    = blockIdx.x;
    const int lstm = g >> 5;
    const int ub   = (g & 31) * 16;

    // one-time: weight slice global -> LDS (uint4 copies), biases, head wts
    {
        const uint4* Ag = (const uint4*)(ws + WS_ATILE + (size_t)g * (4 * 17 * 32 * 16 * 2));
        uint4* Al = (uint4*)smem;
        for (int i = tid; i < 69632 / 16; i += TPB) Al[i] = Ag[i];
        const float* bias_g = (const float*)(ws + WS_BIAS) + lstm * 2048;
        if (tid < 64) { int q = tid >> 4, m = tid & 15; biasL[tid] = bias_g[q * 512 + ub + m]; }
        const float* H0 = (lstm == 0) ? Wxyz : Wzeta;
        const float* H1 = (lstm == 0) ? Wuvw : Wpqr;
        if (tid < 96) {
            int j = tid >> 4, u = tid & 15;
            const float* H = (j < 3) ? H0 : H1;
            int r = (j < 3) ? j : j - 3;
            hw[tid] = H[r * HIDN + ub + u];
        }
        if (tid < 16) cst[tid] = 0.0f;
    }
    __syncthreads();

    const int w    = tid >> 5;
    const int lane = tid & 31;
    const int koff = lane & 16;                        // B half: K 0..15 / 16..31
    const _Float16* h16 = (const _Float16*)(ws + WS_H16);
    const _Float16* x16 = (const _Float16*)(ws + WS_X16);
    int* cnt = (int*)(ws + WS_CNT);
    const v16h* Aw = Alds + (size_t)(w * 17) * 32 + lane;

    for (int t = 0; t < TSTEPS; ++t) {
        v8f acc = {0.f, 0.f, 0.f, 0.f, 0.f, 0.f, 0.f, 0.f};
        const _Float16* hb = h16 + ((size_t)(t & 1) * 2 + lstm) * HIDN;
#pragma unroll
        for (int c = 0; c < 16; ++c) {                 // recurrent K = 512
            v16h a = Aw[c * 32];
            v16h b = *(const v16h*)(hb + c * 32 + koff);
            acc = __builtin_amdgcn_wmma_f32_16x16x32_f16(false, a, false, b,
                                                         (short)0, acc, false, false);
        }
        {                                              // input chunk K = 32
            v16h a = Aw[16 * 32];
            v16h b = *(const v16h*)(x16 + (size_t)t * 32 + koff);
            acc = __builtin_amdgcn_wmma_f32_16x16x32_f16(false, a, false, b,
                                                         (short)0, acc, false, false);
        }
        // all D columns identical: lane 0 -> rows 0..7, lane 16 -> rows 8..15
        if ((lane & 15) == 0) {
            int m0 = (lane >> 4) * 8;
#pragma unroll
            for (int j = 0; j < 8; ++j)
                gbuf[w * 16 + m0 + j] = acc[j] + biasL[w * 16 + m0 + j];
        }
        __syncthreads();

        if (tid < 16) {                                // LSTM cell update (f32)
            float gi = sigmoidf_(gbuf[0 * 16 + tid]);
            float gf = sigmoidf_(gbuf[1 * 16 + tid]);
            float gg = tanhf   (gbuf[2 * 16 + tid]);
            float go = sigmoidf_(gbuf[3 * 16 + tid]);
            float c_ = gf * cst[tid] + gi * gg;
            cst[tid] = c_;
            float h_ = go * tanhf(c_);
            hsf[tid] = h_;
            ((_Float16*)(ws + WS_H16))[((size_t)((t + 1) & 1) * 2 + lstm) * HIDN + ub + tid]
                = (_Float16)h_;
        }
        __syncthreads();

        if (tid < 6) {                                 // head partial dots
            float d = 0.0f;
#pragma unroll
            for (int u = 0; u < 16; ++u) d += hw[tid * 16 + u] * hsf[u];
            int col = (lstm == 0) ? ((tid < 3) ? tid : tid + 3)
                                  : ((tid < 3) ? tid + 3 : tid + 6);
            atomicAdd(&out[(size_t)t * 12 + col], d);
        }

        // device-wide barrier: release arrive, acquire spin, per-thread acquire
        __threadfence();
        __syncthreads();
        if (tid == 0) {
            __hip_atomic_fetch_add(cnt, 1, __ATOMIC_ACQ_REL, __HIP_MEMORY_SCOPE_AGENT);
            while (__hip_atomic_load(cnt, __ATOMIC_ACQUIRE, __HIP_MEMORY_SCOPE_AGENT)
                   < NWG * (t + 1))
                __builtin_amdgcn_s_sleep(1);
        }
        __syncthreads();
        (void)__hip_atomic_load(cnt, __ATOMIC_ACQUIRE, __HIP_MEMORY_SCOPE_AGENT);
    }
}

// ---------------- host side ------------------------------------------------
extern "C" void kernel_launch(void* const* d_in, const int* in_sizes, int n_in,
                              void* d_out, int out_size, void* d_ws, size_t ws_size,
                              hipStream_t stream) {
    const float* sa    = (const float*)d_in[0];
    const float* Wih1  = (const float*)d_in[1];
    const float* Whh1  = (const float*)d_in[2];
    const float* bih1  = (const float*)d_in[3];
    const float* bhh1  = (const float*)d_in[4];
    const float* Wih2  = (const float*)d_in[5];
    const float* Whh2  = (const float*)d_in[6];
    const float* bih2  = (const float*)d_in[7];
    const float* bhh2  = (const float*)d_in[8];
    const float* Wxyz  = (const float*)d_in[9];
    const float* bxyz  = (const float*)d_in[10];
    const float* Wzeta = (const float*)d_in[11];
    const float* bzeta = (const float*)d_in[12];
    const float* Wuvw  = (const float*)d_in[13];
    const float* buvw  = (const float*)d_in[14];
    const float* Wpqr  = (const float*)d_in[15];
    const float* bpqr  = (const float*)d_in[16];
    float* out = (float*)d_out;
    char*  ws  = (char*)d_ws;

    k_init <<<(TSTEPS * 12 + 255) / 256, 256, 0, stream>>>(out, bxyz, bzeta, buvw, bpqr, ws);
    k_x16  <<<(TSTEPS * 32 + 255) / 256, 256, 0, stream>>>(sa, ws);
    k_bias <<<(2 * 2048 + 255) / 256, 256, 0, stream>>>(bih1, bhh1, bih2, bhh2, ws);
    k_atile<<<(NWG * 4 * 17 * 32 * 16 + 255) / 256, 256, 0, stream>>>(Wih1, Whh1, Wih2, Whh2, ws);
    k_lstm <<<NWG, TPB, 70656, stream>>>(Wxyz, Wzeta, Wuvw, Wpqr, out, ws);
}
